// NATCrossLayer_1108101563139
// MI455X (gfx1250) — matrix operand order repaired and checked by reading, not compiled, not run on hardware
//
#include <hip/hip_runtime.h>
#include <math.h>

typedef __bf16 bf16_t;
typedef __attribute__((ext_vector_type(16))) __bf16 v16bf;
typedef __attribute__((ext_vector_type(8)))  float  v8f;

#define DIMC   768
#define HEADS  12
#define DHD    64
#define HID    3072
#define NBATCH 8
#define NSEQ   1024
#define NROWS  (NBATCH * NSEQ)   // 8192

// ---------------------------------------------------------------------------
// Async global->LDS (CDNA5): bypasses VGPRs, tracked by ASYNCcnt.
// Signature discovered via compile probe: (v4i global*, v4i lds*, imm, imm).
// ---------------------------------------------------------------------------
#if defined(__has_builtin)
#if __has_builtin(__builtin_amdgcn_global_load_async_to_lds_b128) && \
    __has_builtin(__builtin_amdgcn_s_wait_asynccnt)
#define HAVE_ASYNC_LDS 1
#endif
#endif
#ifndef HAVE_ASYNC_LDS
#define HAVE_ASYNC_LDS 0
#endif

#if HAVE_ASYNC_LDS
typedef int v4i_g __attribute__((vector_size(16)));
typedef __attribute__((address_space(1))) v4i_g* glb_v4i_p;
typedef __attribute__((address_space(3))) v4i_g* lds_v4i_p;
__device__ __forceinline__ void async_copy_b128(void* lds_dst, const void* gsrc) {
  // Flat LDS address: low 32 bits are the LDS byte offset (ISA 10.2 aperture rule).
  __builtin_amdgcn_global_load_async_to_lds_b128(
      (glb_v4i_p)(unsigned long long)gsrc,
      (lds_v4i_p)(unsigned int)(unsigned long long)lds_dst, 0, 0);
}
__device__ __forceinline__ void wait_async_lds() {
  __builtin_amdgcn_s_wait_asynccnt(0);
}
#endif

// ---------------------------------------------------------------------------
// WMMA helpers (gfx1250, wave32, 16x16x32 bf16 -> f32)
// ---------------------------------------------------------------------------
__device__ __forceinline__ v8f wmma_bf16f32(v16bf a, v16bf b, v8f c) {
  return __builtin_amdgcn_wmma_f32_16x16x32_bf16(false, a, false, b, (short)0, c,
                                                 false, false);
}

// A fragment: 16(M) x 32(K) bf16, row-major (M x K) source, leading dim lda.
// lanes 0-15: M=lane, K 0..15; lanes 16-31: M=lane-16, K 16..31.
// Per-lane reads are two contiguous 16B runs -> 2x ds_load_b128.
__device__ __forceinline__ v16bf load_a_frag(const bf16_t* __restrict__ base, int lda) {
  const int l  = threadIdx.x & 31;
  const int m  = l & 15;
  const int kh = (l < 16) ? 0 : 16;
  const bf16_t* r = base + (size_t)m * lda + kh;
  v16bf a;
#pragma unroll
  for (int e = 0; e < 8; ++e) a[e] = r[e];
#pragma unroll
  for (int e = 0; e < 8; ++e) a[8 + e] = r[8 + e];
  return a;
}

// B fragment: 32(K) x 16(N), from a K-major (N x K row-major) source:
// B[k][n] = src[n*ld + k]. lanes 0-15: N=lane, K=0..15; lanes 16-31: K=16..31.
// Per-lane reads contiguous 32B -> 2x ds_load_b128.
__device__ __forceinline__ v16bf load_bk_frag(const bf16_t* __restrict__ base, int ld) {
  const int l    = threadIdx.x & 31;
  const int n    = l & 15;
  const int koff = (l < 16) ? 0 : 16;
  const bf16_t* r = base + (size_t)n * ld + koff;
  v16bf b;
#pragma unroll
  for (int e = 0; e < 16; ++e) b[e] = r[e];
  return b;
}

// ---------------------------------------------------------------------------
// fp32 -> bf16 transposing weight conversion: wt[n*K + k] = (bf16)w[k*N + n]
// ---------------------------------------------------------------------------
__global__ __launch_bounds__(256) void f2bf_transpose_kernel(
    const float* __restrict__ w, bf16_t* __restrict__ wt, int K, int N) {
  __shared__ float tile[32][33];
  const int ntn = N >> 5;
  const int tn = blockIdx.x % ntn;
  const int tk = blockIdx.x / ntn;
  const int lx = threadIdx.x & 31;
  const int ly = threadIdx.x >> 5;  // 0..7
#pragma unroll
  for (int s = 0; s < 4; ++s) {
    const int kk = ly + 8 * s;
    tile[kk][lx] = w[(size_t)(tk * 32 + kk) * N + tn * 32 + lx];
  }
  __syncthreads();
#pragma unroll
  for (int s = 0; s < 4; ++s) {
    const int nn = ly + 8 * s;
    wt[(size_t)(tn * 32 + nn) * K + tk * 32 + lx] = (bf16_t)tile[lx][nn];
  }
}

// q_extra [B][N][H][D] fp32 -> q_b [B][H][N][D] bf16
__global__ __launch_bounds__(256) void q_permute_kernel(const float* __restrict__ q,
                                                        bf16_t* __restrict__ out) {
  size_t i = (size_t)blockIdx.x * 256 + threadIdx.x;
  const int d = (int)(i & 63);
  size_t t = i >> 6;
  const int n = (int)(t % NSEQ); t /= NSEQ;
  const int hh = (int)(t % HEADS);
  const int bb = (int)(t / HEADS);
  out[i] = (bf16_t)q[(((size_t)bb * NSEQ + n) * HEADS + hh) * DHD + d];
}

// LayerNorm (fp32 in) -> bf16 out. One block per row.
__global__ __launch_bounds__(256) void ln_bf16_kernel(const float* __restrict__ x,
                                                      const float* __restrict__ g,
                                                      const float* __restrict__ b,
                                                      bf16_t* __restrict__ out) {
  const int row = blockIdx.x;
  const float* xr = x + (size_t)row * DIMC;
  float s = 0.f, s2 = 0.f;
  for (int i = threadIdx.x; i < DIMC; i += 256) {
    float v = xr[i]; s += v; s2 += v * v;
  }
#pragma unroll
  for (int m = 16; m >= 1; m >>= 1) { s += __shfl_xor(s, m); s2 += __shfl_xor(s2, m); }
  __shared__ float ss[8], ss2[8];
  if ((threadIdx.x & 31) == 0) { ss[threadIdx.x >> 5] = s; ss2[threadIdx.x >> 5] = s2; }
  __syncthreads();
  s = 0.f; s2 = 0.f;
#pragma unroll
  for (int w = 0; w < 8; ++w) { s += ss[w]; s2 += ss2[w]; }
  const float mu   = s * (1.0f / DIMC);
  const float var  = s2 * (1.0f / DIMC) - mu * mu;
  const float rstd = rsqrtf(var + 1e-5f);
  bf16_t* orow = out + (size_t)row * DIMC;
  for (int i = threadIdx.x; i < DIMC; i += 256)
    orow[i] = (bf16_t)((xr[i] - mu) * rstd * g[i] + b[i]);
}

// ---------------------------------------------------------------------------
// Tiled bf16 GEMM: C[M,N] = A[M,K] @ W[K,N], W supplied K-major (Wt[N,K]).
// Block tile 128x128, K-step 32, 8 waves of 64x32 (4x2 WMMA tiles).
// ---------------------------------------------------------------------------
#define GBM 128
#define GBN 128
#define GBK 32
#define LDT_S (GBK + 16)   // 48 elems -> 96B row pitch (16B aligned, de-banked)

#define MODE_KV      0
#define MODE_OUTPROJ 1
#define MODE_FFN1    2
#define MODE_FFN2    3

__global__ __launch_bounds__(256) void gemm_bf16_kernel(
    const bf16_t* __restrict__ A, const bf16_t* __restrict__ Wt,
    int M, int N, int K, int mode,
    const float* __restrict__ bias, const float* __restrict__ resid,
    float* __restrict__ outf, bf16_t* __restrict__ outb,
    bf16_t* __restrict__ koutb, bf16_t* __restrict__ voutb) {
  __shared__ __align__(16) bf16_t As[GBM][LDT_S];  // 128 M-rows x 32 K
  __shared__ __align__(16) bf16_t Ws[GBN][LDT_S];  // 128 N-rows x 32 K

  const int blockM = blockIdx.y * GBM;
  const int blockN = blockIdx.x * GBN;
  const int tid  = threadIdx.x;
  const int lane = tid & 31;
  const int wid  = tid >> 5;
  const int m0w  = (wid >> 2) * 64;  // 2 waves along M
  const int n0w  = (wid & 3) * 32;   // 4 waves along N

  v8f acc[4][2];
#pragma unroll
  for (int i = 0; i < 4; ++i)
#pragma unroll
    for (int j = 0; j < 2; ++j)
      acc[i][j] = v8f{0.f, 0.f, 0.f, 0.f, 0.f, 0.f, 0.f, 0.f};

  const int ar = tid >> 2;            // 0..63
  const int ac = (tid & 3) * 8;       // 0,8,16,24
  const bf16_t* gA = A  + (size_t)(blockM + ar) * K + ac;
  const bf16_t* gB = Wt + (size_t)(blockN + ar) * K + ac;

  for (int k0 = 0; k0 < K; k0 += GBK) {
    __syncthreads();
#if HAVE_ASYNC_LDS
    async_copy_b128(&As[ar][ac],      gA + k0);
    async_copy_b128(&As[ar + 64][ac], gA + (size_t)64 * K + k0);
    async_copy_b128(&Ws[ar][ac],      gB + k0);
    async_copy_b128(&Ws[ar + 64][ac], gB + (size_t)64 * K + k0);
    wait_async_lds();
#else
    const uint4 t0 = *(const uint4*)(gA + k0);
    const uint4 t1 = *(const uint4*)(gA + (size_t)64 * K + k0);
    const uint4 t2 = *(const uint4*)(gB + k0);
    const uint4 t3 = *(const uint4*)(gB + (size_t)64 * K + k0);
    *(uint4*)&As[ar][ac]      = t0;
    *(uint4*)&As[ar + 64][ac] = t1;
    *(uint4*)&Ws[ar][ac]      = t2;
    *(uint4*)&Ws[ar + 64][ac] = t3;
#endif
    if (k0 + GBK < K) {  // prefetch next K tile
      __builtin_prefetch(gA + k0 + GBK, 0, 0);
      __builtin_prefetch(gB + k0 + GBK, 0, 0);
    }
    __syncthreads();

    v16bf af[4];
#pragma unroll
    for (int i = 0; i < 4; ++i) af[i] = load_a_frag(&As[m0w + 16 * i][0], LDT_S);
    v16bf bfg[2];
#pragma unroll
    for (int j = 0; j < 2; ++j) bfg[j] = load_bk_frag(&Ws[n0w + 16 * j][0], LDT_S);
#pragma unroll
    for (int i = 0; i < 4; ++i)
#pragma unroll
      for (int j = 0; j < 2; ++j)
        acc[i][j] = wmma_bf16f32(af[i], bfg[j], acc[i][j]);
  }

  // Epilogue. D layout: lane -> col (lane&15), VGPR r -> row r + 8*(lane>=16).
  const int colb = lane & 15;
  const int rowb = (lane >> 4) * 8;
#pragma unroll
  for (int i = 0; i < 4; ++i)
#pragma unroll
    for (int j = 0; j < 2; ++j)
#pragma unroll
      for (int r = 0; r < 8; ++r) {
        const int row = blockM + m0w + 16 * i + rowb + r;
        const int col = blockN + n0w + 16 * j + colb;
        const float val = acc[i][j][r];
        if (mode == MODE_KV) {
          // col = sel*768 + h*64 + d;  K -> [B,H,N,D],  V -> [B,H,D,N] (V^T)
          const int bb = row >> 10, n = row & 1023;
          const int sel = col / DIMC;
          const int cc  = col - sel * DIMC;
          const int hh  = cc >> 6, d = cc & 63;
          const bf16_t bv = (bf16_t)val;
          if (sel == 0)
            koutb[(((size_t)bb * HEADS + hh) * NSEQ + n) * DHD + d] = bv;
          else
            voutb[(((size_t)bb * HEADS + hh) * DHD + d) * NSEQ + n] = bv;
        } else if (mode == MODE_FFN1) {
          const float t = val + bias[col];
          const float gl = 0.5f * t * (1.0f + erff(t * 0.70710678118f));  // exact GELU
          outb[(size_t)row * HID + col] = (bf16_t)gl;
        } else {  // MODE_OUTPROJ / MODE_FFN2: bias + residual, fp32 out
          outf[(size_t)row * DIMC + col] =
              val + bias[col] + resid[(size_t)row * DIMC + col];
        }
      }
}

// ---------------------------------------------------------------------------
// Flash attention: one block = 4 waves = 64 query rows of one (b,h).
// K is [B,H,N,64]; V is [B,H,64,N] (transposed by the KV GEMM epilogue) so
// both QK^T and PV B-fragments are contiguous K-major reads from LDS.
// ---------------------------------------------------------------------------
#define LDK_S (DHD + 16)   // 80 elems -> 160B pitch
#define LDV_S (32 + 16)    // 48 elems -> 96B pitch
#define LDP_S (32 + 16)

__global__ __launch_bounds__(128) void attn_kernel(const bf16_t* __restrict__ q,
                                                   const bf16_t* __restrict__ k,
                                                   const bf16_t* __restrict__ vt,
                                                   bf16_t* __restrict__ o) {
  __shared__ __align__(16) bf16_t Ks[32][LDK_S];      // 32 keys x 64 d
  __shared__ __align__(16) bf16_t Vts[DHD][LDV_S];    // 64 d x 32 keys
  __shared__ __align__(16) bf16_t Ps[4][16][LDP_S];   // per-wave P scratch

  const int qtiles = NSEQ / 64;                       // 16
  const int bh = blockIdx.x / qtiles;                 // 0..95
  const int qt = blockIdx.x % qtiles;
  const int b  = bh / HEADS;
  const int h  = bh % HEADS;
  const int wid  = threadIdx.x >> 5;
  const int lane = threadIdx.x & 31;
  const int tid  = threadIdx.x;

  const bf16_t* qb  = q  + (size_t)bh * NSEQ * DHD;
  const bf16_t* kb  = k  + (size_t)bh * NSEQ * DHD;
  const bf16_t* vtb = vt + (size_t)bh * DHD * NSEQ;

  const int q0 = qt * 64 + wid * 16;
  const v16bf aq0 = load_a_frag(qb + (size_t)q0 * DHD, DHD);       // d 0..31
  const v16bf aq1 = load_a_frag(qb + (size_t)q0 * DHD + 32, DHD);  // d 32..63

  v8f oacc[4];
#pragma unroll
  for (int t = 0; t < 4; ++t) oacc[t] = v8f{0.f, 0.f, 0.f, 0.f, 0.f, 0.f, 0.f, 0.f};
  float rmax[8], rsum[8];
#pragma unroll
  for (int r = 0; r < 8; ++r) { rmax[r] = -3.0e38f; rsum[r] = 0.f; }

  for (int c = 0; c < NSEQ / 32; ++c) {
    __syncthreads();
    {  // stage 32-key K chunk (contiguous 4KB) + V^T chunk (64 rows x 64B)
      const bf16_t* gk = kb + (size_t)c * 32 * DHD;
#pragma unroll
      for (int it = 0; it < 2; ++it) {
        const int u = tid + it * 128;                    // 0..255
        const int kr = u >> 3, ks = (u & 7) * 8;         // Ks[32][64]
        const int vr = u >> 2, vs = (u & 3) * 8;         // Vts[64][32]
        const bf16_t* gvp = vtb + (size_t)vr * NSEQ + c * 32 + vs;
#if HAVE_ASYNC_LDS
        async_copy_b128(&Ks[kr][ks], gk + u * 8);
        async_copy_b128(&Vts[vr][vs], gvp);
#else
        const uint4 tk = *(const uint4*)(gk + u * 8);
        const uint4 tv = *(const uint4*)gvp;
        *(uint4*)&Ks[kr][ks]  = tk;
        *(uint4*)&Vts[vr][vs] = tv;
#endif
      }
#if HAVE_ASYNC_LDS
      wait_async_lds();
#endif
    }
    __syncthreads();

    // S = Q @ K^T for 16q x 32keys (d split in two 32-chunks)
    v8f s0 = v8f{0.f, 0.f, 0.f, 0.f, 0.f, 0.f, 0.f, 0.f};
    v8f s1 = v8f{0.f, 0.f, 0.f, 0.f, 0.f, 0.f, 0.f, 0.f};
    s0 = wmma_bf16f32(aq0, load_bk_frag(&Ks[0][0],   LDK_S), s0);
    s0 = wmma_bf16f32(aq1, load_bk_frag(&Ks[0][32],  LDK_S), s0);
    s1 = wmma_bf16f32(aq0, load_bk_frag(&Ks[16][0],  LDK_S), s1);
    s1 = wmma_bf16f32(aq1, load_bk_frag(&Ks[16][32], LDK_S), s1);

    const float scale = 0.125f;  // 64^-0.5
#pragma unroll
    for (int r = 0; r < 8; ++r) {
      const float a0 = s0[r] * scale, a1 = s1[r] * scale;
      float mx = fmaxf(a0, a1);
#pragma unroll
      for (int msk = 8; msk >= 1; msk >>= 1) mx = fmaxf(mx, __shfl_xor(mx, msk));
      const float nm = fmaxf(rmax[r], mx);
      const float alpha = __expf(rmax[r] - nm);
      rmax[r] = nm;
      const float p0 = __expf(a0 - nm);
      const float p1 = __expf(a1 - nm);
      float ls = p0 + p1;
#pragma unroll
      for (int msk = 8; msk >= 1; msk >>= 1) ls += __shfl_xor(ls, msk);
      rsum[r] = rsum[r] * alpha + ls;
#pragma unroll
      for (int t = 0; t < 4; ++t) oacc[t][r] *= alpha;
      // D-layout -> LDS P[16][32] for re-load in A-layout
      const int rr = r + ((lane >> 4) << 3);
      Ps[wid][rr][lane & 15]        = (bf16_t)p0;
      Ps[wid][rr][16 + (lane & 15)] = (bf16_t)p1;
    }

    // O += P @ V  (contraction over 32 keys, 4 d-tiles of 16; V^T is K-major)
    const v16bf ap = load_a_frag(&Ps[wid][0][0], LDP_S);
#pragma unroll
    for (int t = 0; t < 4; ++t)
      oacc[t] = wmma_bf16f32(ap, load_bk_frag(&Vts[16 * t][0], LDV_S), oacc[t]);
  }

  // Finalize: divide by row sums, write [B][N][H*64+d] bf16
#pragma unroll
  for (int t = 0; t < 4; ++t)
#pragma unroll
    for (int r = 0; r < 8; ++r) {
      const int qq = q0 + r + ((lane >> 4) << 3);
      const int d  = 16 * t + (lane & 15);
      const float val = oacc[t][r] / rsum[r];
      o[((size_t)(b * NSEQ + qq)) * DIMC + h * DHD + d] = (bf16_t)val;
    }
}

// ---------------------------------------------------------------------------
// Orchestration
// ---------------------------------------------------------------------------
extern "C" void kernel_launch(void* const* d_in, const int* in_sizes, int n_in,
                              void* d_out, int out_size, void* d_ws, size_t ws_size,
                              hipStream_t stream) {
  (void)in_sizes; (void)n_in; (void)out_size; (void)ws_size;
  const float* x    = (const float*)d_in[0];
  const float* qx   = (const float*)d_in[1];
  const float* ln1g = (const float*)d_in[2];
  const float* ln1b = (const float*)d_in[3];
  const float* wkv  = (const float*)d_in[4];
  const float* wout = (const float*)d_in[5];
  const float* bout = (const float*)d_in[6];
  const float* ln2g = (const float*)d_in[7];
  const float* ln2b = (const float*)d_in[8];
  const float* w1   = (const float*)d_in[9];
  const float* b1   = (const float*)d_in[10];
  const float* w2   = (const float*)d_in[11];
  const float* b2   = (const float*)d_in[12];
  float* out = (float*)d_out;

  char* ws = (char*)d_ws;
  size_t off = 0;
  auto alloc = [&](size_t bytes) -> char* {
    char* p = ws + off;
    off += (bytes + 255) & ~(size_t)255;
    return p;
  };
  bf16_t* wkv_t  = (bf16_t*)alloc((size_t)DIMC * 2 * DIMC * 2);  // [1536][768]
  bf16_t* wout_t = (bf16_t*)alloc((size_t)DIMC * DIMC * 2);      // [768][768]
  bf16_t* w1_t   = (bf16_t*)alloc((size_t)DIMC * HID * 2);       // [3072][768]
  bf16_t* w2_t   = (bf16_t*)alloc((size_t)HID * DIMC * 2);       // [768][3072]
  bf16_t* xn_b   = (bf16_t*)alloc((size_t)NROWS * DIMC * 2);
  bf16_t* q_b    = (bf16_t*)alloc((size_t)NROWS * DIMC * 2);
  bf16_t* k_b    = (bf16_t*)alloc((size_t)NROWS * DIMC * 2);     // [B,H,N,64]
  bf16_t* vT_b   = (bf16_t*)alloc((size_t)NROWS * DIMC * 2);     // [B,H,64,N]
  bf16_t* ao_b   = (bf16_t*)alloc((size_t)NROWS * DIMC * 2);
  float*  xres   = (float*) alloc((size_t)NROWS * DIMC * 4);
  bf16_t* hln_b  = (bf16_t*)alloc((size_t)NROWS * DIMC * 2);
  bf16_t* h1_b   = (bf16_t*)alloc((size_t)NROWS * HID * 2);

  auto tcvt = [&](const float* src, bf16_t* dst, int K, int N) {
    f2bf_transpose_kernel<<<dim3((K / 32) * (N / 32)), dim3(256), 0, stream>>>(
        src, dst, K, N);
  };
  tcvt(wkv,  wkv_t,  DIMC, 2 * DIMC);
  tcvt(wout, wout_t, DIMC, DIMC);
  tcvt(w1,   w1_t,   DIMC, HID);
  tcvt(w2,   w2_t,   HID,  DIMC);

  ln_bf16_kernel<<<dim3(NROWS), dim3(256), 0, stream>>>(x, ln1g, ln1b, xn_b);
  q_permute_kernel<<<dim3(NROWS * DIMC / 256), dim3(256), 0, stream>>>(qx, q_b);

  // kv = ln1(x) @ w_kv -> K [B,H,N,64], V^T [B,H,64,N]
  gemm_bf16_kernel<<<dim3((2 * DIMC) / GBN, NROWS / GBM), dim3(256), 0, stream>>>(
      xn_b, wkv_t, NROWS, 2 * DIMC, DIMC, MODE_KV,
      nullptr, nullptr, nullptr, nullptr, k_b, vT_b);

  attn_kernel<<<dim3(NBATCH * HEADS * (NSEQ / 64)), dim3(128), 0, stream>>>(
      q_b, k_b, vT_b, ao_b);

  // xres = x + attn_out @ w_out + b_out   (fp32)
  gemm_bf16_kernel<<<dim3(DIMC / GBN, NROWS / GBM), dim3(256), 0, stream>>>(
      ao_b, wout_t, NROWS, DIMC, DIMC, MODE_OUTPROJ,
      bout, x, xres, nullptr, nullptr, nullptr);

  ln_bf16_kernel<<<dim3(NROWS), dim3(256), 0, stream>>>(xres, ln2g, ln2b, hln_b);

  // h1 = gelu(ln2 @ w1 + b1)   (bf16)
  gemm_bf16_kernel<<<dim3(HID / GBN, NROWS / GBM), dim3(256), 0, stream>>>(
      hln_b, w1_t, NROWS, HID, DIMC, MODE_FFN1,
      b1, nullptr, nullptr, h1_b, nullptr, nullptr);

  // out = xres + h1 @ w2 + b2   (fp32)
  gemm_bf16_kernel<<<dim3(DIMC / GBN, NROWS / GBM), dim3(256), 0, stream>>>(
      h1_b, w2_t, NROWS, DIMC, HID, MODE_FFN2,
      b2, xres, out, nullptr, nullptr, nullptr);
}
